// MultiHeadAttention_35201551958205
// MI455X (gfx1250) — matrix operand order repaired and checked by reading, not compiled, not run on hardware
//
#include <hip/hip_runtime.h>

#define D_MODEL 768
#define N_HEAD  12
#define DH      64
#define BATCH   4
#define SEQ     2048
#define M_TOTAL (BATCH * SEQ)  // 8192
#define W_ELEMS (D_MODEL * D_MODEL)  // 589824

typedef __attribute__((ext_vector_type(16))) __bf16 v16bf;
typedef __attribute__((ext_vector_type(8)))  float  v8f;

union BfVec {
  v16bf v;
  __bf16 h[16];
  uint4 q[2];
};

__device__ inline v8f wmma_bf16(v16bf a, v16bf b, v8f c) {
  return __builtin_amdgcn_wmma_f32_16x16x32_bf16(false, a, false, b, (short)0, c, false, false);
}

// ---- Fragment loaders (CDNA5 ISA 7.12.2 layouts) ----
// A fragment (16x32, 16-bit): lane holds row (lane&15); K = (j&7) + 16*(j>>3) + 8*(lane>>4)
__device__ inline v16bf load_a_f32(const float* base, int ld, int row0, int k0, int lane) {
  int g = lane >> 4;
  const float* p = base + (size_t)(row0 + (lane & 15)) * ld + k0 + (g << 3);
  float4 x0 = *(const float4*)(p + 0);
  float4 x1 = *(const float4*)(p + 4);
  float4 x2 = *(const float4*)(p + 16);
  float4 x3 = *(const float4*)(p + 20);
  BfVec r;
  r.h[0] = (__bf16)x0.x;  r.h[1] = (__bf16)x0.y;  r.h[2] = (__bf16)x0.z;  r.h[3] = (__bf16)x0.w;
  r.h[4] = (__bf16)x1.x;  r.h[5] = (__bf16)x1.y;  r.h[6] = (__bf16)x1.z;  r.h[7] = (__bf16)x1.w;
  r.h[8] = (__bf16)x2.x;  r.h[9] = (__bf16)x2.y;  r.h[10] = (__bf16)x2.z; r.h[11] = (__bf16)x2.w;
  r.h[12] = (__bf16)x3.x; r.h[13] = (__bf16)x3.y; r.h[14] = (__bf16)x3.z; r.h[15] = (__bf16)x3.w;
  return r.v;
}

__device__ inline v16bf load_a_bf16(const __bf16* base, int ld, int row0, int k0, int lane) {
  int g = lane >> 4;
  const __bf16* p = base + (size_t)(row0 + (lane & 15)) * ld + k0 + (g << 3);
  BfVec r;
  r.q[0] = *(const uint4*)(p);       // K = koff + 0..7
  r.q[1] = *(const uint4*)(p + 16);  // K = koff + 16..23
  return r.v;
}

// B fragment (32x16, 16-bit): lane holds column (lane&15); K = (j&7) + 16*(j>>3) + 8*(lane>>4)
// base is B^T row-major (W[n][k] for X@W^T, Kh[key][dh], or Vt[dh][key])
__device__ inline v16bf load_b_bf16(const __bf16* base, int ld, int n0, int k0, int lane) {
  int g = lane >> 4;
  const __bf16* p = base + (size_t)(n0 + (lane & 15)) * ld + k0 + (g << 3);
  BfVec r;
  r.q[0] = *(const uint4*)(p);       // K = koff + 0..7
  r.q[1] = *(const uint4*)(p + 16);  // K = koff + 16..23
  return r.v;
}

// ==================== Kernel 0: fp32 -> bf16 weight staging ====================
__global__ __launch_bounds__(256) void cvt_bf16_kernel(const float* __restrict__ src,
                                                       __bf16* __restrict__ dst, int n4) {
  int i = blockIdx.x * blockDim.x + threadIdx.x;
  if (i < n4) {
    float4 x = *(const float4*)(src + (size_t)i * 4);
    union { __bf16 h[4]; uint2 u; } r;
    r.h[0] = (__bf16)x.x; r.h[1] = (__bf16)x.y; r.h[2] = (__bf16)x.z; r.h[3] = (__bf16)x.w;
    *(uint2*)(dst + (size_t)i * 4) = r.u;
  }
}

// ==================== Kernel 1: fused QKV projection ====================
// Block 256 thr = 8 waves (4 M x 2 N); wave tile 32x64; block tile 128x128.
// z picks Q/K/V. Q (pre-scaled by 1/8), K -> [b,h,s,dh] bf16 ; V -> [b,h,dh,s] bf16.
__global__ __launch_bounds__(256) void qkv_proj_kernel(
    const float* __restrict__ qx, const float* __restrict__ kx, const float* __restrict__ vx,
    const __bf16* __restrict__ Wqb, const __bf16* __restrict__ Wkb, const __bf16* __restrict__ Wvb,
    const float* __restrict__ bq, const float* __restrict__ bk, const float* __restrict__ bv,
    __bf16* __restrict__ qh, __bf16* __restrict__ kh, __bf16* __restrict__ vt) {
  int lane = threadIdx.x & 31;
  int wave = threadIdx.x >> 5;
  int m0 = blockIdx.y * 128 + (wave >> 1) * 32;
  int n0 = blockIdx.x * 128 + (wave & 1) * 64;
  int z = blockIdx.z;
  const float* X = (z == 0) ? qx : (z == 1) ? kx : vx;
  const __bf16* W = (z == 0) ? Wqb : (z == 1) ? Wkb : Wvb;
  const float* bias = (z == 0) ? bq : (z == 1) ? bk : bv;
  float scl = (z == 0) ? 0.125f : 1.0f;  // fold 1/sqrt(dh) into Q

  v8f acc[2][4] = {};
  for (int kk = 0; kk < D_MODEL; kk += 32) {
    v16bf a0 = load_a_f32(X, D_MODEL, m0, kk, lane);
    v16bf a1 = load_a_f32(X, D_MODEL, m0 + 16, kk, lane);
#pragma unroll
    for (int ni = 0; ni < 4; ++ni) {
      v16bf bfrag = load_b_bf16(W, D_MODEL, n0 + 16 * ni, kk, lane);
      acc[0][ni] = wmma_bf16(a0, bfrag, acc[0][ni]);
      acc[1][ni] = wmma_bf16(a1, bfrag, acc[1][ni]);
    }
  }

  int g = lane >> 4, c = lane & 15;
#pragma unroll
  for (int mi = 0; mi < 2; ++mi) {
#pragma unroll
    for (int ni = 0; ni < 4; ++ni) {
      int n = n0 + ni * 16 + c;
      float bb = bias[n];
      int h = n >> 6, dh = n & 63;
#pragma unroll
      for (int i = 0; i < 8; ++i) {
        int m = m0 + mi * 16 + i + 8 * g;  // C-frag: lanes16-31 hold rows M+8
        int b = m >> 11, s = m & (SEQ - 1);
        __bf16 val = (__bf16)((acc[mi][ni][i] + bb) * scl);
        if (z == 2) {
          vt[((size_t)(b * N_HEAD + h) * DH + dh) * SEQ + s] = val;
        } else {
          __bf16* dst = (z == 0) ? qh : kh;
          dst[((size_t)(b * N_HEAD + h) * SEQ + s) * DH + dh] = val;
        }
      }
    }
  }
}

// ==================== Kernel 2: flash attention ====================
// Block 128 thr = 4 waves; wave owns 16 q-rows x 64 dh; 64-key tiles (16 WMMA/iter).
__global__ __launch_bounds__(128) void attn_kernel(
    const __bf16* __restrict__ qh, const __bf16* __restrict__ kh,
    const __bf16* __restrict__ vt, const int* __restrict__ mask,
    __bf16* __restrict__ y) {
  __shared__ __bf16 pl[4][16][64];  // per-wave P tile: C-layout -> A-layout staging
  int lane = threadIdx.x & 31;
  int wave = threadIdx.x >> 5;
  int bh = blockIdx.y;
  int b = bh / N_HEAD, h = bh % N_HEAD;
  int q0 = blockIdx.x * 64 + wave * 16;
  const __bf16* Qb = qh + (size_t)bh * SEQ * DH;
  const __bf16* Kb = kh + (size_t)bh * SEQ * DH;
  const __bf16* Vb = vt + (size_t)bh * DH * SEQ;
  const int* mb = mask + (size_t)b * SEQ;
  int g = lane >> 4, c = lane & 15;

  // Q A-fragments (already scaled by 1/8) held for the whole key loop
  v16bf qa0 = load_a_bf16(Qb, DH, q0, 0, lane);
  v16bf qa1 = load_a_bf16(Qb, DH, q0, 32, lane);

  v8f O[4] = {};
  float mrow[8], lrow[8];
#pragma unroll
  for (int i = 0; i < 8; ++i) { mrow[i] = 0.0f; lrow[i] = 0.0f; }

  for (int kt = 0; kt < SEQ; kt += 64) {
    // scores: 16(q) x 64(keys); K-dim = dh = 64 -> 2 WMMA per 16-key subtile
    v8f s[4] = {};
#pragma unroll
    for (int t = 0; t < 4; ++t) {
      s[t] = wmma_bf16(qa0, load_b_bf16(Kb, DH, kt + 16 * t, 0, lane), s[t]);
      s[t] = wmma_bf16(qa1, load_b_bf16(Kb, DH, kt + 16 * t, 32, lane), s[t]);
    }

    int msk[4];
#pragma unroll
    for (int t = 0; t < 4; ++t) msk[t] = mb[kt + 16 * t + c];

    float x[4][8];
#pragma unroll
    for (int t = 0; t < 4; ++t)
#pragma unroll
      for (int i = 0; i < 8; ++i) x[t][i] = msk[t] ? s[t][i] : -1e30f;

    float corr[8];
#pragma unroll
    for (int i = 0; i < 8; ++i) {
      // row max across 16 lanes of this half-wave (C-frag cols live in lanes)
      float t0 = fmaxf(fmaxf(x[0][i], x[1][i]), fmaxf(x[2][i], x[3][i]));
      t0 = fmaxf(t0, __shfl_xor(t0, 1, 32));
      t0 = fmaxf(t0, __shfl_xor(t0, 2, 32));
      t0 = fmaxf(t0, __shfl_xor(t0, 4, 32));
      t0 = fmaxf(t0, __shfl_xor(t0, 8, 32));
      float mnew = fmaxf(mrow[i], t0);  // mrow starts at 0 => always valid shift
      corr[i] = __expf(mrow[i] - mnew);
      mrow[i] = mnew;
    }
#pragma unroll
    for (int t = 0; t < 4; ++t)
#pragma unroll
      for (int i = 0; i < 8; ++i) x[t][i] = __expf(x[t][i] - mrow[i]);
#pragma unroll
    for (int i = 0; i < 8; ++i) {
      float r = (x[0][i] + x[1][i]) + (x[2][i] + x[3][i]);
      r += __shfl_xor(r, 1, 32);
      r += __shfl_xor(r, 2, 32);
      r += __shfl_xor(r, 4, 32);
      r += __shfl_xor(r, 8, 32);
      lrow[i] = lrow[i] * corr[i] + r;
    }
#pragma unroll
    for (int f = 0; f < 4; ++f)
#pragma unroll
      for (int i = 0; i < 8; ++i) O[f][i] *= corr[i];

    // P: C-layout -> LDS -> two A-fragments (same-wave DS ops are in-order)
#pragma unroll
    for (int t = 0; t < 4; ++t)
#pragma unroll
      for (int i = 0; i < 8; ++i) pl[wave][i + 8 * g][16 * t + c] = (__bf16)x[t][i];
    __builtin_amdgcn_wave_barrier();
    BfVec pr0, pr1;
    const __bf16* lp0 = &pl[wave][c][g * 8];
    pr0.q[0] = *(const uint4*)(lp0);
    pr0.q[1] = *(const uint4*)(lp0 + 16);
    const __bf16* lp1 = &pl[wave][c][32 + g * 8];
    pr1.q[0] = *(const uint4*)(lp1);
    pr1.q[1] = *(const uint4*)(lp1 + 16);
    __builtin_amdgcn_wave_barrier();

    // O += P @ V ; V as B-fragment from transposed [dh][key] layout
#pragma unroll
    for (int f = 0; f < 4; ++f) {
      O[f] = wmma_bf16(pr0.v, load_b_bf16(Vb, SEQ, f * 16, kt, lane), O[f]);
      O[f] = wmma_bf16(pr1.v, load_b_bf16(Vb, SEQ, f * 16, kt + 32, lane), O[f]);
    }
  }

  // epilogue: y[b][s][h*64+dh] = O / l  (bf16)
#pragma unroll
  for (int f = 0; f < 4; ++f) {
    int d = h * 64 + f * 16 + c;
#pragma unroll
    for (int i = 0; i < 8; ++i) {
      int s = q0 + i + 8 * g;
      y[((size_t)(b * SEQ + s)) * D_MODEL + d] = (__bf16)(O[f][i] / lrow[i]);
    }
  }
}

// ==================== Kernel 3: output projection ====================
__global__ __launch_bounds__(256) void out_proj_kernel(
    const __bf16* __restrict__ y, const __bf16* __restrict__ Wob,
    const float* __restrict__ bo, float* __restrict__ out) {
  int lane = threadIdx.x & 31;
  int wave = threadIdx.x >> 5;
  int m0 = blockIdx.y * 128 + (wave >> 1) * 32;
  int n0 = blockIdx.x * 128 + (wave & 1) * 64;

  v8f acc[2][4] = {};
  for (int kk = 0; kk < D_MODEL; kk += 32) {
    v16bf a0 = load_a_bf16(y, D_MODEL, m0, kk, lane);
    v16bf a1 = load_a_bf16(y, D_MODEL, m0 + 16, kk, lane);
#pragma unroll
    for (int ni = 0; ni < 4; ++ni) {
      v16bf bfrag = load_b_bf16(Wob, D_MODEL, n0 + 16 * ni, kk, lane);
      acc[0][ni] = wmma_bf16(a0, bfrag, acc[0][ni]);
      acc[1][ni] = wmma_bf16(a1, bfrag, acc[1][ni]);
    }
  }

  int g = lane >> 4, c = lane & 15;
#pragma unroll
  for (int mi = 0; mi < 2; ++mi) {
#pragma unroll
    for (int ni = 0; ni < 4; ++ni) {
      int n = n0 + ni * 16 + c;
      float bb = bo[n];
#pragma unroll
      for (int i = 0; i < 8; ++i) {
        int m = m0 + mi * 16 + i + 8 * g;
        out[(size_t)m * D_MODEL + n] = acc[mi][ni][i] + bb;
      }
    }
  }
}

extern "C" void kernel_launch(void* const* d_in, const int* in_sizes, int n_in,
                              void* d_out, int out_size, void* d_ws, size_t ws_size,
                              hipStream_t stream) {
  (void)in_sizes; (void)n_in; (void)out_size; (void)ws_size;
  const float* q   = (const float*)d_in[0];
  const float* k   = (const float*)d_in[1];
  const float* v   = (const float*)d_in[2];
  const int* mask  = (const int*)d_in[3];
  const float* Wq  = (const float*)d_in[4];
  const float* bq  = (const float*)d_in[5];
  const float* Wk  = (const float*)d_in[6];
  const float* bk  = (const float*)d_in[7];
  const float* Wv  = (const float*)d_in[8];
  const float* bv  = (const float*)d_in[9];
  const float* Wo  = (const float*)d_in[10];
  const float* bo  = (const float*)d_in[11];
  float* out = (float*)d_out;

  size_t elems = (size_t)M_TOTAL * D_MODEL;  // 6,291,456 per activation buffer
  __bf16* Wqb = (__bf16*)d_ws;
  __bf16* Wkb = Wqb + W_ELEMS;
  __bf16* Wvb = Wkb + W_ELEMS;
  __bf16* Wob = Wvb + W_ELEMS;
  __bf16* qh  = Wob + W_ELEMS;
  __bf16* kh  = qh + elems;
  __bf16* vt  = kh + elems;
  __bf16* yy  = vt + elems;  // total ws: ~52.5 MB

  int n4 = W_ELEMS / 4;
  int cblk = (n4 + 255) / 256;
  cvt_bf16_kernel<<<cblk, 256, 0, stream>>>(Wq, Wqb, n4);
  cvt_bf16_kernel<<<cblk, 256, 0, stream>>>(Wk, Wkb, n4);
  cvt_bf16_kernel<<<cblk, 256, 0, stream>>>(Wv, Wvb, n4);
  cvt_bf16_kernel<<<cblk, 256, 0, stream>>>(Wo, Wob, n4);

  qkv_proj_kernel<<<dim3(D_MODEL / 128, M_TOTAL / 128, 3), 256, 0, stream>>>(
      q, k, v, Wqb, Wkb, Wvb, bq, bk, bv, qh, kh, vt);
  attn_kernel<<<dim3(SEQ / 64, BATCH * N_HEAD), 128, 0, stream>>>(qh, kh, vt, mask, yy);
  out_proj_kernel<<<dim3(D_MODEL / 128, M_TOTAL / 128), 256, 0, stream>>>(yy, Wob, bo, out);
}
